// CustomMultiheadAttention_10350871183602
// MI455X (gfx1250) — matrix operand order repaired and checked by reading, not compiled, not run on hardware
//
#include <hip/hip_runtime.h>
#include <hip/hip_bf16.h>

typedef __bf16 bf16;
typedef __attribute__((ext_vector_type(8)))  bf16  v8bf;
typedef __attribute__((ext_vector_type(16))) bf16  v16bf;
typedef __attribute__((ext_vector_type(8)))  float v8f;

#define SEQ    2048
#define DMODEL 1024
#define NHEAD  16
#define HDIM   64
#define BATCH  4
#define MROWS  (BATCH * SEQ)

// ---------------------------------------------------------------------------
// WMMA helpers (CDNA5 wave32, 16x16x32 bf16 -> f32)
// ---------------------------------------------------------------------------
static __device__ inline v8f wmma_bf16(v16bf a, v16bf b, v8f c) {
  return __builtin_amdgcn_wmma_f32_16x16x32_bf16(
      /*neg_a=*/false, a, /*neg_b=*/false, b,
      /*c_mod=*/(short)0, c, /*reuse_a=*/false, /*reuse_b=*/false);
}

static __device__ inline v16bf join8(v8bf lo, v8bf hi) {
  return __builtin_shufflevector(lo, hi, 0,1,2,3,4,5,6,7,8,9,10,11,12,13,14,15);
}

// A-fragment (16x32 bf16, row-major, leading dim ld):
// lane L: row = L&15 ; elems[0..15] = K (lh*16)+0..15
static __device__ inline v16bf load_a_bf16(const bf16* base, int ld, int lm, int lh) {
  const bf16* p = base + (size_t)lm * ld + lh * 16;
  return *(const v16bf*)p;
}

// B-fragment (32x16 bf16). Source holds B^T row-major (row n = 16 contiguous K).
// lane n (0-15): K 0..15 ; lane n+16: K 16..31  -> one 32B contiguous read/lane.
static __device__ inline v16bf load_b_bf16(const bf16* base, int ld, int lm, int lh) {
  const bf16* p = base + (size_t)lm * ld + lh * 16;
  return *(const v16bf*)p;  // 2x global_load_b128
}

// ---------------------------------------------------------------------------
// fp32 -> bf16 conversion (grid-stride free: exact-size launches, n % 8 == 0)
// ---------------------------------------------------------------------------
__global__ void cvt_f32_bf16(const float* __restrict__ in, bf16* __restrict__ out, int n) {
  int i = (blockIdx.x * blockDim.x + threadIdx.x) * 8;
  if (i + 7 < n) {
    float4 f0 = *(const float4*)(in + i);
    float4 f1 = *(const float4*)(in + i + 4);
    out[i]     = (bf16)f0.x; out[i + 1] = (bf16)f0.y;
    out[i + 2] = (bf16)f0.z; out[i + 3] = (bf16)f0.w;
    out[i + 4] = (bf16)f1.x; out[i + 5] = (bf16)f1.y;
    out[i + 6] = (bf16)f1.z; out[i + 7] = (bf16)f1.w;
  }
}

// ---------------------------------------------------------------------------
// GEMM: Y[m,n] = sum_k A[m,k] * W[n,k] + bias[n]   (A @ W^T + b), all-bf16 in.
// mode 0: bf16 out, [B,H,S,Hd]  (Q,K)
// mode 1: bf16 out, [B,H,Hd,S]  (V transposed for P.V B-fragments)
// mode 2: fp32 out, [M,1024]    (final output projection -> d_out)
// Block: 256 thr = 8 waves (4 M x 2 N), block tile 128x128, per-wave 32x64,
// K-step 32 -> 8 v_wmma + 6 fragment loads per step.
// ---------------------------------------------------------------------------
__global__ __launch_bounds__(256) void gemm_qkvo(
    const bf16* __restrict__ A, const bf16* __restrict__ W,
    const float* __restrict__ bias, void* __restrict__ outp, int mode) {
  const int lane = threadIdx.x & 31, wid = threadIdx.x >> 5;
  const int lm = lane & 15, lh = lane >> 4;
  const int wm = wid & 3, wn = wid >> 2;
  const int mW = blockIdx.x * 128 + wm * 32;
  const int nW = blockIdx.y * 128 + wn * 64;

  v8f acc[2][4] = {};
  for (int k0 = 0; k0 < DMODEL; k0 += 32) {
    const bf16* abase = A + (size_t)mW * DMODEL + k0;
    const bf16* wbase = W + (size_t)nW * DMODEL + k0;
    if (k0 + 32 < DMODEL) {  // pull next K tile toward the WGP
      __builtin_prefetch(abase + 32 + (size_t)lm * DMODEL, 0, 3);
      __builtin_prefetch(wbase + 32 + (size_t)lm * DMODEL, 0, 3);
    }
    v16bf a0 = load_a_bf16(abase, DMODEL, lm, lh);
    v16bf a1 = load_a_bf16(abase + (size_t)16 * DMODEL, DMODEL, lm, lh);
    v16bf b0 = load_b_bf16(wbase, DMODEL, lm, lh);
    v16bf b1 = load_b_bf16(wbase + (size_t)16 * DMODEL, DMODEL, lm, lh);
    v16bf b2 = load_b_bf16(wbase + (size_t)32 * DMODEL, DMODEL, lm, lh);
    v16bf b3 = load_b_bf16(wbase + (size_t)48 * DMODEL, DMODEL, lm, lh);
    acc[0][0] = wmma_bf16(a0, b0, acc[0][0]);
    acc[0][1] = wmma_bf16(a0, b1, acc[0][1]);
    acc[0][2] = wmma_bf16(a0, b2, acc[0][2]);
    acc[0][3] = wmma_bf16(a0, b3, acc[0][3]);
    acc[1][0] = wmma_bf16(a1, b0, acc[1][0]);
    acc[1][1] = wmma_bf16(a1, b1, acc[1][1]);
    acc[1][2] = wmma_bf16(a1, b2, acc[1][2]);
    acc[1][3] = wmma_bf16(a1, b3, acc[1][3]);
  }
  // Epilogue. C layout: vgpr i, lane l -> m = i + 8*(l>=16), n = l&15.
  for (int bc = 0; bc < 4; ++bc) {
    const int n = nW + bc * 16 + lm;
    const float bv = bias[n];
    const int h = n >> 6, hd = n & 63;
    for (int ar = 0; ar < 2; ++ar) {
      for (int i = 0; i < 8; ++i) {
        const int m = mW + ar * 16 + i + 8 * lh;
        const float v = acc[ar][bc][i] + bv;
        const int b = m >> 11, s = m & (SEQ - 1);
        if (mode == 0) {
          ((bf16*)outp)[(((size_t)b * NHEAD + h) * SEQ + s) * HDIM + hd] = (bf16)v;
        } else if (mode == 1) {
          ((bf16*)outp)[(((size_t)b * NHEAD + h) * HDIM + hd) * SEQ + s] = (bf16)v;
        } else {
          ((float*)outp)[(size_t)m * DMODEL + n] = v;
        }
      }
    }
  }
}

// ---------------------------------------------------------------------------
// Flash-attention forward. One wave owns 16 query rows; 8 waves per block
// cover 128 rows of one (b,h). 64 keys per iteration:
//   S(16x64) = Q K^T -> 8 wmma ; online softmax (fp32, cross-lane shfl per
//   16-lane half matching the C layout) ; P repacked via per-wave LDS into two
//   16x32 A-fragments (in-wave ordering via s_wait_dscnt) ; O += P V -> 8 wmma.
// Output written bf16 [B,S,D] for direct consumption by the final projection.
// ---------------------------------------------------------------------------
__global__ __launch_bounds__(256) void attn_fwd(
    const bf16* __restrict__ Q,   // [B,H,S,Hd]
    const bf16* __restrict__ K,   // [B,H,S,Hd]
    const bf16* __restrict__ Vt,  // [B,H,Hd,S]
    bf16* __restrict__ O) {       // [B,S,D] bf16
  __shared__ __align__(32) bf16 pls[8][16 * 64];  // per-wave P scratch (2 KB)
  const int lane = threadIdx.x & 31, wid = threadIdx.x >> 5;
  const int lm = lane & 15, lh = lane >> 4;
  const int bh = blockIdx.x >> 4;      // b*16+h
  const int chunk = blockIdx.x & 15;   // S/128 chunks
  const int q0 = chunk * 128 + wid * 16;
  const int b = bh >> 4, h = bh & 15;

  const bf16* qb = Q + (size_t)bh * SEQ * HDIM;
  const bf16* kb = K + (size_t)bh * SEQ * HDIM;
  const bf16* vb = Vt + (size_t)bh * HDIM * SEQ;

  const v16bf qa0 = load_a_bf16(qb + (size_t)q0 * HDIM + 0,  HDIM, lm, lh);
  const v16bf qa1 = load_a_bf16(qb + (size_t)q0 * HDIM + 32, HDIM, lm, lh);

  float mx[8], ls[8], r[8];
  v8f o[4] = {};
  for (int i = 0; i < 8; ++i) { mx[i] = -1e30f; ls[i] = 0.f; }
  bf16* pl = pls[wid];

  for (int kv = 0; kv < SEQ; kv += 64) {
    // ---- scores for 64 keys: 4 fragments of 16 keys, Hd split in two K-steps
    v8f s[4];
    for (int j = 0; j < 4; ++j) {
      const bf16* kj = kb + (size_t)(kv + j * 16) * HDIM;
      v16bf blo = load_b_bf16(kj, HDIM, lm, lh);        // d 0..31
      v16bf bhi = load_b_bf16(kj + 32, HDIM, lm, lh);   // d 32..63
      v8f t = {};
      t = wmma_bf16(qa0, blo, t);
      s[j] = wmma_bf16(qa1, bhi, t);
    }
    // ---- online softmax per row (row m = i + 8*lh, cols = lane&15 per frag)
    const float scale = 0.125f;  // 1/sqrt(64)
    for (int i = 0; i < 8; ++i) {
      float p0 = s[0][i] * scale, p1 = s[1][i] * scale;
      float p2 = s[2][i] * scale, p3 = s[3][i] * scale;
      float rm = fmaxf(fmaxf(p0, p1), fmaxf(p2, p3));
      rm = fmaxf(rm, __shfl_xor(rm, 1, 32));
      rm = fmaxf(rm, __shfl_xor(rm, 2, 32));
      rm = fmaxf(rm, __shfl_xor(rm, 4, 32));
      rm = fmaxf(rm, __shfl_xor(rm, 8, 32));
      const float mnew = fmaxf(mx[i], rm);
      r[i] = __expf(mx[i] - mnew);
      mx[i] = mnew;
      p0 = __expf(p0 - mnew); p1 = __expf(p1 - mnew);
      p2 = __expf(p2 - mnew); p3 = __expf(p3 - mnew);
      const int row = i + 8 * lh;
      pl[row * 64 +  0 + lm] = (bf16)p0;
      pl[row * 64 + 16 + lm] = (bf16)p1;
      pl[row * 64 + 32 + lm] = (bf16)p2;
      pl[row * 64 + 48 + lm] = (bf16)p3;
      float rs = (p0 + p1) + (p2 + p3);
      rs += __shfl_xor(rs, 1, 32); rs += __shfl_xor(rs, 2, 32);
      rs += __shfl_xor(rs, 4, 32); rs += __shfl_xor(rs, 8, 32);
      ls[i] = ls[i] * r[i] + rs;
    }
    for (int c = 0; c < 4; ++c)
      for (int i = 0; i < 8; ++i) o[c][i] *= r[i];

    // per-lane ds stores must land before cross-lane A-frag reads; only this
    // wave touches its LDS slice -> DS-counter wait suffices (no barrier).
    asm volatile("s_wait_dscnt 0x0" ::: "memory");
    const v16bf pa0 = load_a_bf16(pl,      64, lm, lh);  // keys kv+0..31
    const v16bf pa1 = load_a_bf16(pl + 32, 64, lm, lh);  // keys kv+32..63

    for (int c = 0; c < 4; ++c) {
      const bf16* vc = vb + (size_t)(c * 16) * SEQ + kv;
      v16bf vf0 = load_b_bf16(vc,      SEQ, lm, lh);
      v16bf vf1 = load_b_bf16(vc + 32, SEQ, lm, lh);
      o[c] = wmma_bf16(pa0, vf0, o[c]);
      o[c] = wmma_bf16(pa1, vf1, o[c]);
    }
  }

  for (int c = 0; c < 4; ++c) {
    for (int i = 0; i < 8; ++i) {
      const int srow = q0 + i + 8 * lh;
      const int col = h * HDIM + c * 16 + lm;
      O[((size_t)(b * SEQ + srow)) * DMODEL + col] = (bf16)(o[c][i] / ls[i]);
    }
  }
}

// ---------------------------------------------------------------------------
extern "C" void kernel_launch(void* const* d_in, const int* in_sizes, int n_in,
                              void* d_out, int out_size, void* d_ws, size_t ws_size,
                              hipStream_t stream) {
  (void)in_sizes; (void)n_in; (void)out_size; (void)ws_size;
  const float* query = (const float*)d_in[0];
  const float* key_i = (const float*)d_in[1];
  const float* value = (const float*)d_in[2];
  const float* Wq = (const float*)d_in[3];
  const float* bq = (const float*)d_in[4];
  const float* Wk = (const float*)d_in[5];
  const float* bk = (const float*)d_in[6];
  const float* Wv = (const float*)d_in[7];
  const float* bv = (const float*)d_in[8];
  const float* Wo = (const float*)d_in[9];
  const float* bo = (const float*)d_in[10];

  char* ws = (char*)d_ws;
  const size_t ACT_ELEMS = (size_t)MROWS * DMODEL;          // 8.4M
  const size_t ACT_BYTES = ACT_ELEMS * sizeof(bf16);        // 16 MB
  const size_t W_ELEMS   = (size_t)DMODEL * DMODEL;         // 1M
  // [0..3): Q,K,Vt head-split bf16 ; [3): attn out bf16 ; [4..7): x_bf ; then weights
  bf16* q_bf  = (bf16*)(ws + 0 * ACT_BYTES);
  bf16* k_bf  = (bf16*)(ws + 1 * ACT_BYTES);
  bf16* vt_bf = (bf16*)(ws + 2 * ACT_BYTES);
  bf16* at_bf = (bf16*)(ws + 3 * ACT_BYTES);
  bf16* xq_bf = (bf16*)(ws + 4 * ACT_BYTES);
  bf16* xk_bf = (bf16*)(ws + 5 * ACT_BYTES);
  bf16* xv_bf = (bf16*)(ws + 6 * ACT_BYTES);
  bf16* wq_bf = (bf16*)(ws + 7 * ACT_BYTES);
  bf16* wk_bf = wq_bf + W_ELEMS;
  bf16* wv_bf = wk_bf + W_ELEMS;
  bf16* wo_bf = wv_bf + W_ELEMS;

  const int ACT_BLKS = (int)(ACT_ELEMS / (256 * 8));  // 4096
  const int W_BLKS   = (int)(W_ELEMS / (256 * 8));    // 512
  cvt_f32_bf16<<<ACT_BLKS, 256, 0, stream>>>(query, xq_bf, (int)ACT_ELEMS);
  cvt_f32_bf16<<<ACT_BLKS, 256, 0, stream>>>(key_i, xk_bf, (int)ACT_ELEMS);
  cvt_f32_bf16<<<ACT_BLKS, 256, 0, stream>>>(value, xv_bf, (int)ACT_ELEMS);
  cvt_f32_bf16<<<W_BLKS, 256, 0, stream>>>(Wq, wq_bf, (int)W_ELEMS);
  cvt_f32_bf16<<<W_BLKS, 256, 0, stream>>>(Wk, wk_bf, (int)W_ELEMS);
  cvt_f32_bf16<<<W_BLKS, 256, 0, stream>>>(Wv, wv_bf, (int)W_ELEMS);
  cvt_f32_bf16<<<W_BLKS, 256, 0, stream>>>(Wo, wo_bf, (int)W_ELEMS);

  dim3 gg(MROWS / 128, DMODEL / 128), gb(256);
  gemm_qkvo<<<gg, gb, 0, stream>>>(xq_bf, wq_bf, bq, (void*)q_bf, 0);
  gemm_qkvo<<<gg, gb, 0, stream>>>(xk_bf, wk_bf, bk, (void*)k_bf, 0);
  gemm_qkvo<<<gg, gb, 0, stream>>>(xv_bf, wv_bf, bv, (void*)vt_bf, 1);

  attn_fwd<<<BATCH * NHEAD * (SEQ / 128), 256, 0, stream>>>(q_bf, k_bf, vt_bf, at_bf);

  gemm_qkvo<<<gg, gb, 0, stream>>>(at_bf, wo_bf, bo, d_out, 2);
}
